// Net1_29188597744302
// MI455X (gfx1250) — compile-verified, matrix operand order checked by reading, not run on hardware
//
#include <hip/hip_runtime.h>
#include <hip/hip_bf16.h>
#include <math.h>

// Problem constants (match reference)
#define B_      32
#define T_      2048
#define H_      1024
#define I_      4
#define O_      4
#define ALPHA   0.05f      // DT/TAU = 0.5/10
#define GGAIN   1.5f

// Kernel geometry
#define NW      16         // persistent workgroups (each owns H_/NW rows of G)
#define SLICE   64         // H_/NW
#define THREADS 256        // 8 wave32 waves
#define WAVES   8
#define HP      1032       // padded LDS row (bf16 elems): 2064B stride -> conflict-free b128 reads

typedef __bf16 bf16_t;
typedef __attribute__((ext_vector_type(16))) __bf16 v16bf;
typedef __attribute__((ext_vector_type(8)))  float  v8f;

union Frag { v16bf v; float4 f4[2]; };

// Hardware tanh (V_TANH_F32) on gfx1250; libm fallback otherwise.
#if __has_builtin(__builtin_amdgcn_tanhf)
  #define FAST_TANH(x) __builtin_amdgcn_tanhf(x)
#elif __has_builtin(__builtin_amdgcn_tanh_f32)
  #define FAST_TANH(x) __builtin_amdgcn_tanh_f32(x)
#else
  #define FAST_TANH(x) tanhf(x)
#endif

// Workspace layout (bytes)
#define WS_RBUF   0                                   // 2 * B_*H_ bf16   = 131072
#define WS_GB     (WS_RBUF + 2 * B_ * H_ * 2)         // H_*H_ bf16       = 2097152
#define WS_POUT   (WS_GB + H_ * H_ * 2)               // 2*NW*B_*O_ float = 16384
#define WS_BAR    (WS_POUT + 2 * NW * B_ * O_ * 4)    // 1 uint
#define SMEM_BYTES (SLICE * HP * 2 + WAVES * 16 * O_ * 4)

// ---------------------------------------------------------------------------
// Prep: G fp32 -> bf16 (row-major [n][k] == B^T layout WMMA wants),
//       r[0] slab = tanh(h0) broadcast over batch, zero the barrier counter.
// ---------------------------------------------------------------------------
__global__ void prep_kernel(const float* __restrict__ G, const float* __restrict__ h0,
                            bf16_t* __restrict__ Gb, bf16_t* __restrict__ r0,
                            unsigned* __restrict__ bar) {
    const int gid = blockIdx.x * THREADS + threadIdx.x;   // 0 .. H_*H_/4 - 1
    const float4 g4 = *(const float4*)(G + (size_t)gid * 4);
    union { bf16_t h[4]; uint2 u; } p;
    p.h[0] = (bf16_t)g4.x; p.h[1] = (bf16_t)g4.y;
    p.h[2] = (bf16_t)g4.z; p.h[3] = (bf16_t)g4.w;
    *(uint2*)(Gb + (size_t)gid * 4) = p.u;
    if (gid < B_ * H_) {
        const int k = gid & (H_ - 1);
        r0[gid] = (bf16_t)FAST_TANH(h0[k]);
    }
    if (gid == 0) *bar = 0u;
}

// ---------------------------------------------------------------------------
// Persistent RNN scan. Grid = NW blocks x 256 threads, all co-resident.
// Wave w: M-tile = w&1 (batch rows), N-tile = w>>1 (16 hidden cols of slice).
// ---------------------------------------------------------------------------
__global__ void __launch_bounds__(THREADS, 1)
rnn_scan_kernel(const float* __restrict__ x, const float* __restrict__ Wi,
                const float* __restrict__ Ww, const float* __restrict__ h0,
                float* __restrict__ out, bf16_t* __restrict__ rbuf,
                const bf16_t* __restrict__ Gb, float* __restrict__ pout,
                unsigned* __restrict__ bar) {
    extern __shared__ char smem[];
    bf16_t (*Gs)[HP]       = (bf16_t (*)[HP])smem;                      // 64 x 1032 bf16
    float  (*wpo)[16][O_]  = (float (*)[16][O_])(smem + SLICE * HP * 2);

    const int tid   = threadIdx.x;
    const int wg    = blockIdx.x;
    const int wave  = tid >> 5;
    const int lane  = tid & 31;
    const int hlf   = lane >> 4;           // 0: lanes 0-15, 1: lanes 16-31
    const int lq    = lane & 15;
    const int mtile = wave & 1;            // batch rows 0-15 / 16-31
    const int ntile = wave >> 1;           // 0..3 -> 16 hidden cols each
    const int n0    = wg * SLICE;
    const int hcol  = n0 + ntile * 16 + lq;       // this lane's hidden column (C N-index)
    const int nrow  = ntile * 16 + lq;            // G row inside LDS slice (B fragment)
    const int half8 = hlf * 8;                    // K sub-offset per WMMA bf16 layout
    const int ab    = mtile * 16 + lq;            // batch row this lane loads for A

    // Stage G slice into LDS (bf16, padded rows)
    for (int c = tid; c < SLICE * (H_ / 8); c += THREADS) {
        const int row = c >> 7;
        const int col = (c & 127) * 8;
        *(float4*)&Gs[row][col] = *(const float4*)(Gb + (size_t)(n0 + row) * H_ + col);
    }

    // Per-lane constants: input/output projection rows for this hidden column
    const float4 wi = *(const float4*)(Wi + (size_t)hcol * I_);
    const float4 ww = *(const float4*)(Ww + (size_t)hcol * O_);
    float hreg[8];
    const float h0v = h0[hcol];
    #pragma unroll
    for (int g = 0; g < 8; ++g) hreg[g] = h0v;    // h state lives in registers

    __syncthreads();
    const bf16_t* gr = &Gs[nrow][0];
    const int osel = lq & 3;                      // output index this lane finalizes

    for (int t = 0; t < T_; ++t) {
        // Fold previous step's partial outputs into d_out (one designated WG)
        if (t > 0 && wg == ((t - 1) & (NW - 1)) && tid < B_ * O_) {
            const float* ps = pout + (size_t)((t - 1) & 1) * NW * (B_ * O_);
            float s = 0.f;
            #pragma unroll
            for (int j = 0; j < NW; ++j) s += ps[j * (B_ * O_) + tid];
            const int b = tid >> 2, o = tid & 3;
            out[(size_t)b * T_ * O_ + (size_t)(t - 1) * O_ + o] = s;
        }

        const bf16_t* rcur = rbuf + (size_t)(t & 1) * (B_ * H_);
        bf16_t*       rnxt = rbuf + (size_t)((t + 1) & 1) * (B_ * H_);
        const bf16_t* ar   = rcur + (size_t)ab * H_;

        // ---- r_t @ G^T tile: two independent WMMA chains over K=1024 ----
        v8f acc0 = {0.f,0.f,0.f,0.f,0.f,0.f,0.f,0.f};
        v8f acc1 = {0.f,0.f,0.f,0.f,0.f,0.f,0.f,0.f};
        #pragma unroll 4
        for (int k0 = 0; k0 < H_ / 2; k0 += 32) {
            Frag a0, b0, a1, b1;
            a0.f4[0] = *(const float4*)(ar + k0 + half8);
            a0.f4[1] = *(const float4*)(ar + k0 + 16 + half8);
            b0.f4[0] = *(const float4*)(gr + k0 + half8);
            b0.f4[1] = *(const float4*)(gr + k0 + 16 + half8);
            a1.f4[0] = *(const float4*)(ar + (H_/2) + k0 + half8);
            a1.f4[1] = *(const float4*)(ar + (H_/2) + k0 + 16 + half8);
            b1.f4[0] = *(const float4*)(gr + (H_/2) + k0 + half8);
            b1.f4[1] = *(const float4*)(gr + (H_/2) + k0 + 16 + half8);
            acc0 = __builtin_amdgcn_wmma_f32_16x16x32_bf16(false, a0.v, false, b0.v,
                                                           (short)0, acc0, false, false);
            acc1 = __builtin_amdgcn_wmma_f32_16x16x32_bf16(false, a1.v, false, b1.v,
                                                           (short)0, acc1, false, false);
        }
        const v8f acc = acc0 + acc1;

        // ---- state update + hw tanh + bf16 r store + O=4 projection partials ----
        float po[8][4];
        #pragma unroll
        for (int g = 0; g < 8; ++g) {
            const int b = mtile * 16 + hlf * 8 + g;            // C layout: M = g (+8 for hi half)
            const float4 xb = *(const float4*)(x + ((size_t)b * T_ + t) * I_);
            const float xwi = xb.x*wi.x + xb.y*wi.y + xb.z*wi.z + xb.w*wi.w;
            const float h  = hreg[g];
            const float hn = h + ALPHA * (GGAIN * acc[g] + xwi - h);
            hreg[g] = hn;
            const float rv = FAST_TANH(hn);
            rnxt[(size_t)b * H_ + hcol] = (bf16_t)rv;
            po[g][0] = rv * ww.x; po[g][1] = rv * ww.y;
            po[g][2] = rv * ww.z; po[g][3] = rv * ww.w;
        }
        __builtin_prefetch(x + ((size_t)ab * T_ + (t + 1 < T_ ? t + 1 : t)) * I_, 0, 1);

        // ---- two-phase cross-lane reduce (stays within 16-lane halves) ----
        // Phase 1: reduce all 32 (g,o) values over 4-lane groups.
        #pragma unroll
        for (int g = 0; g < 8; ++g) {
            #pragma unroll
            for (int o = 0; o < 4; ++o) {
                float v = po[g][o];
                v += __shfl_xor(v, 1, 32);
                v += __shfl_xor(v, 2, 32);
                po[g][o] = v;
            }
        }
        // Phase 2: each lane finalizes o = lq&3 across the 4 groups.
        float vg[8];
        #pragma unroll
        for (int g = 0; g < 8; ++g) {
            float v = (osel == 0) ? po[g][0] : (osel == 1) ? po[g][1]
                    : (osel == 2) ? po[g][2] : po[g][3];
            v += __shfl_xor(v, 4, 32);
            v += __shfl_xor(v, 8, 32);
            vg[g] = v;
        }
        if (lq < 4) {   // lanes 0-3 / 16-19 publish the 4 outputs for their 8 rows
            #pragma unroll
            for (int g = 0; g < 8; ++g)
                wpo[wave][hlf * 8 + g][osel] = vg[g];
        }
        __syncthreads();
        if (tid < B_ * O_) {   // combine the 4 waves sharing each M-tile -> per-WG partial
            const int b = tid >> 2, o = tid & 3, m = b & 15, first = b >> 4;
            const float s = wpo[first][m][o] + wpo[first + 2][m][o] +
                            wpo[first + 4][m][o] + wpo[first + 6][m][o];
            pout[((size_t)(t & 1) * NW + wg) * (B_ * O_) + tid] = s;
        }

        // ---- device-wide barrier (monotonic counter, release/acquire) ----
        __builtin_amdgcn_fence(__ATOMIC_RELEASE, "agent");
        __syncthreads();
        if (tid == 0) {
            __hip_atomic_fetch_add(bar, 1u, __ATOMIC_RELEASE, __HIP_MEMORY_SCOPE_AGENT);
            const unsigned target = (unsigned)(t + 1) * NW;
            while (__hip_atomic_load(bar, __ATOMIC_ACQUIRE, __HIP_MEMORY_SCOPE_AGENT) < target)
                __builtin_amdgcn_s_sleep(1);
        }
        __syncthreads();
        __builtin_amdgcn_fence(__ATOMIC_ACQUIRE, "agent");
    }

    // Fold the final step's partials
    if (wg == ((T_ - 1) & (NW - 1)) && tid < B_ * O_) {
        const float* ps = pout + (size_t)((T_ - 1) & 1) * NW * (B_ * O_);
        float s = 0.f;
        #pragma unroll
        for (int j = 0; j < NW; ++j) s += ps[j * (B_ * O_) + tid];
        const int b = tid >> 2, o = tid & 3;
        out[(size_t)b * T_ * O_ + (size_t)(T_ - 1) * O_ + o] = s;
    }
}

// ---------------------------------------------------------------------------
extern "C" void kernel_launch(void* const* d_in, const int* in_sizes, int n_in,
                              void* d_out, int out_size, void* d_ws, size_t ws_size,
                              hipStream_t stream) {
    (void)in_sizes; (void)n_in; (void)out_size; (void)ws_size;
    const float* x  = (const float*)d_in[0];   // [B,T,I]
    const float* G  = (const float*)d_in[1];   // [H,H]
    const float* Wi = (const float*)d_in[2];   // [H,I]
    const float* Ww = (const float*)d_in[3];   // [H,O]
    const float* h0 = (const float*)d_in[4];   // [H]
    float* out = (float*)d_out;                // [B,T,O]

    char* ws = (char*)d_ws;
    bf16_t*   rbuf = (bf16_t*)(ws + WS_RBUF);
    bf16_t*   Gb   = (bf16_t*)(ws + WS_GB);
    float*    pout = (float*)(ws + WS_POUT);
    unsigned* bar  = (unsigned*)(ws + WS_BAR);

    // Allow >64KB dynamic LDS (idempotent; ignore errors)
    (void)hipFuncSetAttribute((const void*)rnn_scan_kernel,
                              hipFuncAttributeMaxDynamicSharedMemorySize, SMEM_BYTES);

    prep_kernel<<<(H_ * H_ / 4) / THREADS, THREADS, 0, stream>>>(G, h0, Gb, rbuf, bar);
    rnn_scan_kernel<<<NW, THREADS, SMEM_BYTES, stream>>>(x, Wi, Ww, h0, out,
                                                         rbuf, Gb, pout, bar);
}
